// FasterRCNNFromScratch_33303176413669
// MI455X (gfx1250) — matrix operand order repaired and checked by reading, not compiled, not run on hardware
//
#include <hip/hip_runtime.h>
#include <hip/hip_bf16.h>
#include <math.h>

typedef __attribute__((ext_vector_type(16))) _Float16 v16h;
typedef __attribute__((ext_vector_type(8)))  _Float16 v8h;
typedef __attribute__((ext_vector_type(8)))  float    v8f;

#define IMGF 640.0f
#define NMS_THR 0.7f
#define BBOX_CLAMP 4.135166556742356f
#define PRE_TOPK 1000
#define NPROP 4000
#define POST_TOPK 150

// ---------------------------------------------------------------- staging
// NCHW f32 -> padded NHWC f16 [(H+2)][(W+2)][256] with zero halo
__global__ void k_feat_pad(const float* __restrict__ src, _Float16* __restrict__ dst,
                           int H, int W) {
  int i = blockIdx.x * blockDim.x + threadIdx.x;
  int Wp = W + 2;
  int total = (H + 2) * Wp * 256;
  if (i >= total) return;
  int c = i & 255;
  int x = (i >> 8) % Wp;
  int y = (i >> 8) / Wp;
  _Float16 v = (_Float16)0.0f;
  if (y >= 1 && y <= H && x >= 1 && x <= W)
    v = (_Float16)src[c * H * W + (y - 1) * W + (x - 1)];
  dst[i] = v;
}

// conv weight [co][ci][3][3] -> WMMA-fragment order:
// dst[((tn*72 + kc)*32 + lane)*16 + e], ci = (kc&7)*32 + 16*(lane/16) + e, co = tn*16 + lane%16
__global__ void k_convw_pack(const float* __restrict__ w, _Float16* __restrict__ cp) {
  int i = blockIdx.x * blockDim.x + threadIdx.x;
  if (i >= 16 * 72 * 32 * 16) return;
  int e = i & 15;
  int lane = (i >> 4) & 31;
  int rest = i >> 9;              // tn*72 + kc
  int tn = rest / 72, kc = rest % 72;
  int tap = kc >> 3;
  int ci = (kc & 7) * 32 + ((lane >> 4) << 4) + e;
  int co = tn * 16 + (lane & 15);
  cp[i] = (_Float16)w[(co * 256 + ci) * 9 + tap];
}

// fc weight [N][K] f32 -> fragment order bp[((tn*nkc + kc)*32 + lane)*16 + e]
__global__ void k_pack_b(const float* __restrict__ w, _Float16* __restrict__ bp, int N, int K) {
  int i = blockIdx.x * blockDim.x + threadIdx.x;
  if (i >= N * K) return;
  int e = i & 15;
  int lane = (i >> 4) & 31;
  int rest = i >> 9;
  int nkc = K >> 5;
  int tn = rest / nkc, kc = rest % nkc;
  int col = tn * 16 + (lane & 15);
  int k = kc * 32 + ((lane >> 4) << 4) + e;
  bp[i] = (_Float16)w[col * K + k];
}

// fused cls(21)+bbox(84) head weights (N=112 padded) in fragment order
__global__ void k_headw_pack(const float* __restrict__ cls_w, const float* __restrict__ bbox_w,
                             _Float16* __restrict__ hp) {
  int i = blockIdx.x * blockDim.x + threadIdx.x;
  const int K = 1024, nkc = K >> 5;
  if (i >= 112 * K) return;
  int e = i & 15;
  int lane = (i >> 4) & 31;
  int rest = i >> 9;
  int tn = rest / nkc, kc = rest % nkc;
  int col = tn * 16 + (lane & 15);
  int k = kc * 32 + ((lane >> 4) << 4) + e;
  float v = 0.0f;
  if (col < 21) v = cls_w[col * K + k];
  else if (col < 105) v = bbox_w[(col - 21) * K + k];
  hp[i] = (_Float16)v;
}

__global__ void k_headb(const float* __restrict__ cls_b, const float* __restrict__ bbox_b,
                        float* __restrict__ hb) {
  int j = blockIdx.x * blockDim.x + threadIdx.x;
  if (j < 112) hb[j] = (j < 21) ? cls_b[j] : (j < 105 ? bbox_b[j - 21] : 0.0f);
}

// ------------------------------------------------ 3x3 conv implicit GEMM (WMMA, 16x64/wave)
// featP: padded NHWC f16, wP: packed fragments, hid out: [co][H*W] f16 (bias+ReLU)
__global__ void k_conv3x3(const _Float16* __restrict__ featP, const _Float16* __restrict__ wP,
                          const float* __restrict__ bias, _Float16* __restrict__ hid,
                          int H, int W) {
  const int HW = H * W, Wp = W + 2;
  int lane = threadIdx.x, g = lane >> 4, m = lane & 15;
  int p = blockIdx.x * 16 + m;
  int py = p / W, px = p % W;
  v8f acc[4] = {v8f{}, v8f{}, v8f{}, v8f{}};
  for (int kc = 0; kc < 72; ++kc) {          // K = 9 taps * 256 ci, chunks of 32
    int tap = kc >> 3;
    int dy = tap / 3 - 1, dx = tap % 3 - 1;
    int ci0 = (kc & 7) * 32;
    const _Float16* fp = featP + ((size_t)(py + dy + 1) * Wp + (px + dx + 1)) * 256
                               + ci0 + (g << 4);
    __builtin_prefetch(fp + 256, 0, 1);
    v8h a0 = *(const v8h*)fp;                // K = ci0 + 16g + 0..7
    v8h a1 = *(const v8h*)(fp + 8);          // K = ci0 + 16g + 8..15
    v16h a = __builtin_shufflevector(a0, a1, 0, 1, 2, 3, 4, 5, 6, 7,
                                             8, 9, 10, 11, 12, 13, 14, 15);
#pragma unroll
    for (int t = 0; t < 4; ++t) {
      v16h b = *(const v16h*)(wP + ((size_t)((blockIdx.y * 4 + t) * 72 + kc) * 32 + lane) * 16);
      acc[t] = __builtin_amdgcn_wmma_f32_16x16x32_f16(false, a, false, b, (short)0,
                                                      acc[t], false, false);
    }
  }
#pragma unroll
  for (int t = 0; t < 4; ++t) {
    int co = blockIdx.y * 64 + t * 16 + m;
    float bi = bias[co];
    v8h o;
#pragma unroll
    for (int r = 0; r < 8; ++r) {            // D: M = 8g + r (contiguous in [co][HW])
      float v = acc[t][r] + bi;
      o[r] = (_Float16)(v > 0.0f ? v : 0.0f);
    }
    *(v8h*)(hid + (size_t)co * HW + blockIdx.x * 16 + (g << 3)) = o;
  }
}

// ------------------------------------------------ 1x1 obj/delta convs (thin -> VALU)
__global__ void k_rpn_head(const _Float16* __restrict__ hid,
                           const float* __restrict__ objw, const float* __restrict__ objb,
                           const float* __restrict__ dltw, const float* __restrict__ dltb,
                           float* __restrict__ scores, float* __restrict__ deltas, int HW) {
  int i = blockIdx.x * blockDim.x + threadIdx.x;
  if (i >= HW * 15) return;
  int oc = i / HW;
  int p  = i % HW;
  const float* w; float b;
  if (oc < 3) { w = objw + oc * 256; b = objb[oc]; }
  else        { w = dltw + (oc - 3) * 256; b = dltb[oc - 3]; }
  float s = b;
  for (int ci = 0; ci < 256; ++ci) s += (float)hid[ci * HW + p] * w[ci];
  if (oc < 3) scores[p * 3 + oc] = s;
  else { int c = oc - 3; deltas[(p * 3 + c / 4) * 4 + (c % 4)] = s; }
}

// ------------------------------------------------ bitonic sort (score desc, idx asc)
__global__ void k_sort_init(const float* __restrict__ s, float* __restrict__ key,
                            int* __restrict__ val, int n, int np2) {
  int i = blockIdx.x * blockDim.x + threadIdx.x;
  if (i >= np2) return;
  key[i] = (i < n) ? s[i] : -INFINITY;
  val[i] = i;
}

__device__ __forceinline__ bool comes_first(float ka, int va, float kb, int vb) {
  return (ka > kb) || (ka == kb && va < vb);
}

__global__ void k_bitonic(float* __restrict__ key, int* __restrict__ val, int n, int k, int j) {
  int i = blockIdx.x * blockDim.x + threadIdx.x;
  if (i >= n) return;
  int ixj = i ^ j;
  if (ixj <= i) return;
  bool up = ((i & k) == 0);
  float ki = key[i], kx = key[ixj];
  int   vi = val[i], vx = val[ixj];
  bool first = comes_first(ki, vi, kx, vx);
  if (up ? !first : first) {
    key[i] = kx; key[ixj] = ki;
    val[i] = vx; val[ixj] = vi;
  }
}

// ------------------------------------------------ per-level top-1000 decode
__global__ void k_select_decode(const float* __restrict__ key, const int* __restrict__ val,
                                const float* __restrict__ deltas,
                                float* __restrict__ boxes_all, float* __restrict__ scores_all,
                                int* __restrict__ lvl_all,
                                int W, float stride, float asize, int lvl) {
  int j = blockIdx.x * blockDim.x + threadIdx.x;
  if (j >= PRE_TOPK) return;
  int idx = val[j];
  float s = key[j];
  int p = idx / 3, a = idx % 3;
  int gy = p / W, gx = p % W;
  float sx = (gx + 0.5f) * stride, sy = (gy + 0.5f) * stride;
  const float ars[3] = {0.5f, 1.0f, 2.0f};
  float ar = ars[a];
  float aw = sqrtf(asize * asize / ar);
  float ah = ar * aw;
  float w = fmaxf(aw, 1e-6f), h = fmaxf(ah, 1e-6f);
  float d0 = deltas[idx * 4 + 0], d1 = deltas[idx * 4 + 1];
  float dw = fminf(deltas[idx * 4 + 2], BBOX_CLAMP);
  float dh = fminf(deltas[idx * 4 + 3], BBOX_CLAMP);
  float pcx = sx + d0 * w, pcy = sy + d1 * h;
  float pw = w * expf(dw), ph = h * expf(dh);
  float x1 = fminf(fmaxf(pcx - 0.5f * pw, 0.0f), IMGF - 1.0f);
  float y1 = fminf(fmaxf(pcy - 0.5f * ph, 0.0f), IMGF - 1.0f);
  float x2 = fminf(fmaxf(pcx + 0.5f * pw, 0.0f), IMGF - 1.0f);
  float y2 = fminf(fmaxf(pcy + 0.5f * ph, 0.0f), IMGF - 1.0f);
  bool valid = ((x2 - x1) >= 4.0f) && ((y2 - y1) >= 4.0f);
  int o = lvl * PRE_TOPK + j;
  boxes_all[o * 4 + 0] = x1; boxes_all[o * 4 + 1] = y1;
  boxes_all[o * 4 + 2] = x2; boxes_all[o * 4 + 3] = y2;
  scores_all[o] = valid ? s : -1e30f;
  lvl_all[o] = lvl;
}

__global__ void k_gather(const float* __restrict__ key2, const int* __restrict__ val2,
                         const float* __restrict__ boxes_all, const int* __restrict__ lvl_all,
                         float* __restrict__ boxes_s, float* __restrict__ scores_s,
                         float* __restrict__ offb) {
  int j = blockIdx.x * blockDim.x + threadIdx.x;
  if (j >= NPROP) return;
  int o = val2[j];
  float off = (float)lvl_all[o] * ((IMGF + 1.0f) * 2.0f);
  for (int t = 0; t < 4; ++t) {
    float v = boxes_all[o * 4 + t];
    boxes_s[j * 4 + t] = v;
    offb[j * 4 + t] = v + off;
  }
  scores_s[j] = key2[j];
}

// ------------------------------------------------ greedy NMS + top-150 + level assign
__global__ void k_nms(const float* __restrict__ offb, const float* __restrict__ boxes_s,
                      const float* __restrict__ scores_s,
                      float* __restrict__ props, int* __restrict__ klvl) {
  __shared__ unsigned char keep[NPROP];
  int tid = threadIdx.x;
  for (int j = tid; j < NPROP; j += blockDim.x) keep[j] = 1;
  __syncthreads();
  for (int i = 0; i < NPROP - 1; ++i) {
    if (keep[i]) {
      float ax1 = offb[i * 4 + 0], ay1 = offb[i * 4 + 1];
      float ax2 = offb[i * 4 + 2], ay2 = offb[i * 4 + 3];
      float areai = fmaxf(ax2 - ax1, 0.0f) * fmaxf(ay2 - ay1, 0.0f);
      for (int j = i + 1 + tid; j < NPROP; j += blockDim.x) {
        if (!keep[j]) continue;
        float bx1 = offb[j * 4 + 0], by1 = offb[j * 4 + 1];
        float bx2 = offb[j * 4 + 2], by2 = offb[j * 4 + 3];
        float areaj = fmaxf(bx2 - bx1, 0.0f) * fmaxf(by2 - by1, 0.0f);
        float iw = fmaxf(fminf(ax2, bx2) - fmaxf(ax1, bx1), 0.0f);
        float ih = fmaxf(fminf(ay2, by2) - fmaxf(ay1, by1), 0.0f);
        float inter = iw * ih;
        float iou = inter / fmaxf(areai + areaj - inter, 1e-6f);
        if (iou > NMS_THR) keep[j] = 0;
      }
    }
    __syncthreads();
  }
  if (tid == 0) {
    int cnt = 0;
    for (int pass = 0; pass < 2 && cnt < POST_TOPK; ++pass) {
      for (int j = 0; j < NPROP && cnt < POST_TOPK; ++j) {
        bool live = keep[j] && (scores_s[j] > -1e30f);
        if ((pass == 0) ? live : !live) {
          float x1 = boxes_s[j * 4 + 0], y1 = boxes_s[j * 4 + 1];
          float x2 = boxes_s[j * 4 + 2], y2 = boxes_s[j * 4 + 3];
          props[cnt * 4 + 0] = x1; props[cnt * 4 + 1] = y1;
          props[cnt * 4 + 2] = x2; props[cnt * 4 + 3] = y2;
          float pw = fmaxf(x2 - x1, 0.0f), ph = fmaxf(y2 - y1, 0.0f);
          float k = floorf(4.0f + log2f(sqrtf(pw * ph) / 224.0f + 1e-6f));
          k = fminf(fmaxf(k, 2.0f), 5.0f);
          klvl[cnt] = (int)k;
          ++cnt;
        }
      }
    }
  }
}

// ------------------------------------------------ ROI-align -> f16 GEMM input [160][12544]
__global__ void k_roialign(const float* __restrict__ p2, const float* __restrict__ p3,
                           const float* __restrict__ p4, const float* __restrict__ p5,
                           const float* __restrict__ props, const int* __restrict__ klvl,
                           _Float16* __restrict__ x16) {
  int i = blockIdx.x * blockDim.x + threadIdx.x;
  const int per = 256 * 49;
  if (i >= 160 * per) return;
  int n = i / per, r = i % per;
  if (n >= POST_TOPK) { x16[i] = (_Float16)0.0f; return; }
  int c = r / 49, yy = (r % 49) / 7, xx = r % 7;
  int l = klvl[n] - 2;
  const float* feat = (l == 0) ? p2 : (l == 1) ? p3 : (l == 2) ? p4 : p5;
  int H = 160 >> l, W = 160 >> l;
  float scale = 1.0f / (float)(4 << l);
  float bx1 = props[n * 4 + 0] * scale, by1 = props[n * 4 + 1] * scale;
  float bx2 = props[n * 4 + 2] * scale, by2 = props[n * 4 + 3] * scale;
  float bw = fmaxf(bx2 - bx1, 1.0f), bh = fmaxf(by2 - by1, 1.0f);
  float X = bx1 + ((xx + 0.5f) / 7.0f) * bw;
  float Y = by1 + ((yy + 0.5f) / 7.0f) * bh;
  float x0f = floorf(X), y0f = floorf(Y);
  float lx = X - x0f, ly = Y - y0f;
  int x0 = (int)fminf(fmaxf(x0f,        0.0f), (float)(W - 1));
  int x1 = (int)fminf(fmaxf(x0f + 1.0f, 0.0f), (float)(W - 1));
  int y0 = (int)fminf(fmaxf(y0f,        0.0f), (float)(H - 1));
  int y1 = (int)fminf(fmaxf(y0f + 1.0f, 0.0f), (float)(H - 1));
  const float* F = feat + c * H * W;
  float v = F[y0 * W + x0] * (1.0f - ly) * (1.0f - lx)
          + F[y0 * W + x1] * (1.0f - ly) * lx
          + F[y1 * W + x0] * ly * (1.0f - lx)
          + F[y1 * W + x1] * ly * lx;
  x16[i] = (_Float16)v;
}

// ------------------------------------------------ WMMA GEMM 16x64/wave, packed B
__global__ void k_gemm_f16v(const _Float16* __restrict__ A, const _Float16* __restrict__ Bp,
                            const float* __restrict__ bias, _Float16* __restrict__ C,
                            int N, int K, int Mvalid, int relu) {
  int tilesN4 = N >> 6;
  int tm = blockIdx.x / tilesN4, t4 = blockIdx.x % tilesN4;
  int lane = threadIdx.x, g = lane >> 4, m = lane & 15;
  int rowA = tm * 16 + m;
  const _Float16* Arow = A + (size_t)rowA * K;
  int nkc = K >> 5;
  v8f acc[4] = {v8f{}, v8f{}, v8f{}, v8f{}};
  for (int kc = 0; kc < nkc; ++kc) {
    const _Float16* ap = Arow + kc * 32 + (g << 4);
    __builtin_prefetch(ap + 64, 0, 1);
    v8h a0 = *(const v8h*)ap;
    v8h a1 = *(const v8h*)(ap + 8);
    v16h a = __builtin_shufflevector(a0, a1, 0, 1, 2, 3, 4, 5, 6, 7,
                                             8, 9, 10, 11, 12, 13, 14, 15);
#pragma unroll
    for (int t = 0; t < 4; ++t) {
      v16h b = *(const v16h*)(Bp + ((size_t)((t4 * 4 + t) * nkc + kc) * 32 + lane) * 16);
      acc[t] = __builtin_amdgcn_wmma_f32_16x16x32_f16(false, a, false, b, (short)0,
                                                      acc[t], false, false);
    }
  }
#pragma unroll
  for (int t = 0; t < 4; ++t) {
    int col = t4 * 64 + t * 16 + m;
    float bi = bias[col];
#pragma unroll
    for (int r = 0; r < 8; ++r) {
      int row = tm * 16 + (g << 3) + r;
      float v = acc[t][r] + bi;
      if (relu) v = fmaxf(v, 0.0f);
      if (row >= Mvalid) v = 0.0f;
      C[(size_t)row * N + col] = (_Float16)v;
    }
  }
}

// head GEMM: N=112 padded (21 cls + 84 bbox), packed B, f32 out [150][105]
__global__ void k_gemm_head(const _Float16* __restrict__ A, const _Float16* __restrict__ Bp,
                            const float* __restrict__ bias, float* __restrict__ out) {
  const int K = 1024, nkc = K >> 5;
  int tm = blockIdx.x / 7, tn = blockIdx.x % 7;
  int lane = threadIdx.x, g = lane >> 4, m = lane & 15;
  int rowA = tm * 16 + m;
  const _Float16* Arow = A + (size_t)rowA * K;
  v8f acc = {};
  for (int kc = 0; kc < nkc; ++kc) {
    const _Float16* ap = Arow + kc * 32 + (g << 4);
    v8h a0 = *(const v8h*)ap;
    v8h a1 = *(const v8h*)(ap + 8);
    v16h a = __builtin_shufflevector(a0, a1, 0, 1, 2, 3, 4, 5, 6, 7,
                                             8, 9, 10, 11, 12, 13, 14, 15);
    v16h b = *(const v16h*)(Bp + ((size_t)(tn * nkc + kc) * 32 + lane) * 16);
    acc = __builtin_amdgcn_wmma_f32_16x16x32_f16(false, a, false, b, (short)0, acc, false, false);
  }
#pragma unroll
  for (int r = 0; r < 8; ++r) {
    int row = tm * 16 + (g << 3) + r;
    int col = tn * 16 + m;
    if (row < POST_TOPK && col < 105) out[row * 105 + col] = acc[r] + bias[col];
  }
}

// ================================================================ host
static inline unsigned cdiv(unsigned a, unsigned b) { return (a + b - 1) / b; }

extern "C" void kernel_launch(void* const* d_in, const int* in_sizes, int n_in,
                              void* d_out, int out_size, void* d_ws, size_t ws_size,
                              hipStream_t stream) {
  (void)in_sizes; (void)n_in; (void)out_size; (void)ws_size;
  const float* pin[4] = {(const float*)d_in[0], (const float*)d_in[1],
                         (const float*)d_in[2], (const float*)d_in[3]};
  const float* rpnw = (const float*)d_in[4];
  const float* rpnb = (const float*)d_in[5];
  const float* objw = (const float*)d_in[6];
  const float* objb = (const float*)d_in[7];
  const float* dltw = (const float*)d_in[8];
  const float* dltb = (const float*)d_in[9];
  const float* fc1w = (const float*)d_in[10];
  const float* fc1b = (const float*)d_in[11];
  const float* fc2w = (const float*)d_in[12];
  const float* fc2b = (const float*)d_in[13];
  const float* clsw = (const float*)d_in[14];
  const float* clsb = (const float*)d_in[15];
  const float* bbw  = (const float*)d_in[16];
  const float* bbb  = (const float*)d_in[17];

  const int Hd[4]      = {160, 80, 40, 20};
  const float strd[4]  = {4.0f, 8.0f, 16.0f, 32.0f};
  const float asz[4]   = {32.0f, 64.0f, 128.0f, 256.0f};
  int HW[4], n3[4];
  for (int l = 0; l < 4; ++l) { HW[l] = Hd[l] * Hd[l]; n3[l] = HW[l] * 3; }

  char* ws = (char*)d_ws;
  size_t off = 0;
  auto alloc = [&](size_t bytes) -> void* {
    off = (off + 255) & ~(size_t)255;
    void* r = ws + off;
    off += bytes;
    return r;
  };

  _Float16 *featP[4], *hid16[4];
  float *scoresL[4], *deltasL[4];
  for (int l = 0; l < 4; ++l) {
    featP[l] = (_Float16*)alloc((size_t)(Hd[l] + 2) * (Hd[l] + 2) * 256 * 2);
    hid16[l] = (_Float16*)alloc((size_t)256 * HW[l] * 2);
    scoresL[l] = (float*)alloc((size_t)n3[l] * 4);
    deltasL[l] = (float*)alloc((size_t)n3[l] * 16);
  }
  _Float16* convP = (_Float16*)alloc((size_t)16 * 72 * 32 * 16 * 2);
  float* sKey = (float*)alloc((size_t)131072 * 4);
  int*   sVal = (int*)  alloc((size_t)131072 * 4);
  float* boxes_all  = (float*)alloc((size_t)NPROP * 16);
  float* scores_all = (float*)alloc((size_t)4096 * 4);
  int*   lvl_all    = (int*)  alloc((size_t)NPROP * 4);
  float* key2 = (float*)alloc((size_t)4096 * 4);
  int*   val2 = (int*)  alloc((size_t)4096 * 4);
  float* boxes_s  = (float*)alloc((size_t)NPROP * 16);
  float* offb     = (float*)alloc((size_t)NPROP * 16);
  float* scores_s = (float*)alloc((size_t)NPROP * 4);
  float* props = (float*)alloc((size_t)POST_TOPK * 16);
  int*   klvl  = (int*)  alloc((size_t)POST_TOPK * 4);
  _Float16* x16   = (_Float16*)alloc((size_t)160 * 12544 * 2);
  _Float16* fc1P  = (_Float16*)alloc((size_t)12544 * 1024 * 2);
  _Float16* fc2P  = (_Float16*)alloc((size_t)1024 * 1024 * 2);
  _Float16* act1  = (_Float16*)alloc((size_t)160 * 1024 * 2);
  _Float16* act2  = (_Float16*)alloc((size_t)160 * 1024 * 2);
  _Float16* headP = (_Float16*)alloc((size_t)112 * 1024 * 2);
  float*    headB = (float*)alloc((size_t)112 * 4);

  // 1) staging: padded NHWC features + packed conv weights
  for (int l = 0; l < 4; ++l) {
    int n = (Hd[l] + 2) * (Hd[l] + 2) * 256;
    k_feat_pad<<<cdiv(n, 256), 256, 0, stream>>>(pin[l], featP[l], Hd[l], Hd[l]);
  }
  k_convw_pack<<<cdiv(16 * 72 * 32 * 16, 256), 256, 0, stream>>>(rpnw, convP);

  // 2) RPN 3x3 conv (WMMA) + 3) 1x1 heads
  for (int l = 0; l < 4; ++l) {
    dim3 grid(HW[l] / 16, 4);
    k_conv3x3<<<grid, 32, 0, stream>>>(featP[l], convP, rpnb, hid16[l], Hd[l], Hd[l]);
    k_rpn_head<<<cdiv(HW[l] * 15, 256), 256, 0, stream>>>(hid16[l], objw, objb, dltw, dltb,
                                                          scoresL[l], deltasL[l], HW[l]);
  }

  // 4) per-level top-1000 (bitonic full sort, stable tie-break) + decode
  for (int l = 0; l < 4; ++l) {
    int np2 = 1;
    while (np2 < n3[l]) np2 <<= 1;
    k_sort_init<<<np2 / 256, 256, 0, stream>>>(scoresL[l], sKey, sVal, n3[l], np2);
    for (int k = 2; k <= np2; k <<= 1)
      for (int j = k >> 1; j > 0; j >>= 1)
        k_bitonic<<<np2 / 256, 256, 0, stream>>>(sKey, sVal, np2, k, j);
    k_select_decode<<<cdiv(PRE_TOPK, 256), 256, 0, stream>>>(sKey, sVal, deltasL[l],
                                                             boxes_all, scores_all, lvl_all,
                                                             Hd[l], strd[l], asz[l], l);
  }

  // 5) global sort of 4000 proposals + NMS + top-150
  k_sort_init<<<4096 / 256, 256, 0, stream>>>(scores_all, key2, val2, NPROP, 4096);
  for (int k = 2; k <= 4096; k <<= 1)
    for (int j = k >> 1; j > 0; j >>= 1)
      k_bitonic<<<4096 / 256, 256, 0, stream>>>(key2, val2, 4096, k, j);
  k_gather<<<cdiv(NPROP, 256), 256, 0, stream>>>(key2, val2, boxes_all, lvl_all,
                                                 boxes_s, scores_s, offb);
  k_nms<<<1, 1024, 0, stream>>>(offb, boxes_s, scores_s, props, klvl);

  // 6) ROI-align -> f16 [160][12544] (rows >= 150 zeroed)
  k_roialign<<<cdiv(160 * 256 * 49, 256), 256, 0, stream>>>(pin[0], pin[1], pin[2], pin[3],
                                                            props, klvl, x16);

  // 7) FC head (WMMA, packed weights)
  k_pack_b<<<cdiv(1024 * 12544, 256), 256, 0, stream>>>(fc1w, fc1P, 1024, 12544);
  k_pack_b<<<cdiv(1024 * 1024, 256), 256, 0, stream>>>(fc2w, fc2P, 1024, 1024);
  k_headw_pack<<<cdiv(112 * 1024, 256), 256, 0, stream>>>(clsw, bbw, headP);
  k_headb<<<1, 128, 0, stream>>>(clsb, bbb, headB);

  k_gemm_f16v<<<(160 / 16) * (1024 / 64), 32, 0, stream>>>(x16, fc1P, fc1b, act1,
                                                           1024, 12544, POST_TOPK, 1);
  k_gemm_f16v<<<(160 / 16) * (1024 / 64), 32, 0, stream>>>(act1, fc2P, fc2b, act2,
                                                           1024, 1024, POST_TOPK, 1);
  k_gemm_head<<<(160 / 16) * 7, 32, 0, stream>>>(act2, headP, headB, (float*)d_out);
}
